// MultiHeadAttentionBlock_87995289960918
// MI455X (gfx1250) — matrix operand order repaired and checked by reading, not compiled, not run on hardware
//
#include <hip/hip_runtime.h>
#include <hip/hip_bf16.h>
#include <stdint.h>

typedef _Float16 f16;
typedef __attribute__((ext_vector_type(16))) _Float16 v16h;
typedef __attribute__((ext_vector_type(8)))  float    v8f;

union V16H { v16h v; uint4 q[2]; };

#define NBATCH 4
#define NSEQ   2048
#define NCH    1536
#define NHEAD  8
#define NQD    128
#define NKD    128
#define NVD    192
#define NNT    1344     /* 8*128 + 128 + 192 */
#define NHV    1536     /* 8*192 */
#define NM     8192     /* NBATCH*NSEQ */
#define BN_EPS_F 1e-6f
#define LN_EPS_F 1e-5f
#define ATTN_SCL 0.08838834764831845f  /* 1/sqrt(128) */
#define LN_ROPE  9.210340371976184f    /* ln(10000) */

__device__ __forceinline__ float wave_sum32(float x) {
#pragma unroll
  for (int off = 16; off >= 1; off >>= 1) x += __shfl_xor(x, off, 32);
  return x;
}

__device__ __forceinline__ v8f wmma_f16(const V16H& a, const V16H& b, v8f c) {
  return __builtin_amdgcn_wmma_f32_16x16x32_f16(false, a.v, false, b.v,
                                                (short)0, c, false, false);
}

// ---------------- Kernel 1: input batchnorm -> f16 ----------------
__global__ __launch_bounds__(256) void bn_in_f16_kernel(
    const float* __restrict__ x, const float* __restrict__ g,
    const float* __restrict__ b, const float* __restrict__ r,
    f16* __restrict__ xn)
{
  size_t i4 = ((size_t)blockIdx.x * blockDim.x + threadIdx.x) * 4;
  if (i4 >= (size_t)NM * NCH) return;
  float4 xv = *(const float4*)(x + i4);
  float vals[4] = {xv.x, xv.y, xv.z, xv.w};
  int c = (int)(i4 % NCH);
#pragma unroll
  for (int k = 0; k < 4; ++k) {
    float sc = rsqrtf(r[c + k] + BN_EPS_F) * g[c + k];
    xn[i4 + k] = (f16)(vals[k] * sc + b[c + k]);
  }
}

// ------------- Kernel 2a: fused QKV weight transpose -> f16 -------------
__global__ __launch_bounds__(256) void wqkv_t_kernel(
    const float* __restrict__ Wq, const float* __restrict__ Wk,
    const float* __restrict__ Wv, f16* __restrict__ WT)
{
  const size_t idx = (size_t)blockIdx.x * 256 + threadIdx.x;
  if (idx >= (size_t)NCH * NNT) return;
  const int n = (int)(idx % NNT);
  const int k = (int)(idx / NNT);
  float v;
  if (n < NHEAD * NQD)              v = Wq[(size_t)n * NCH + k];
  else if (n < NHEAD * NQD + NKD)   v = Wk[(size_t)(n - NHEAD * NQD) * NCH + k];
  else                              v = Wv[(size_t)(n - NHEAD * NQD - NKD) * NCH + k];
  WT[idx] = (f16)v;
}

// ------------- Kernel 2b: output weight transpose -> f16 -------------
__global__ __launch_bounds__(256) void wo_t_kernel(
    const float* __restrict__ Wo, f16* __restrict__ WoT)
{
  const size_t idx = (size_t)blockIdx.x * 256 + threadIdx.x;
  if (idx >= (size_t)NHV * NCH) return;
  const int c = (int)(idx % NCH);
  const int j = (int)(idx / NCH);
  WoT[idx] = (f16)Wo[(size_t)c * NHV + j];  // WoT[j][c] = Wo[c][j]
}

// ---------------- Kernel 3/6: WMMA GEMM  C = A(f16,MxK) * B(f16,KxN) ----------------
// Block tile 256(M) x 64(N): 8 waves stacked in M, each wave computes 32x64.
// Per k-step of 32: 2 A-frags + 4 B-frags (4x b128 A loads + 64 per-element f16 B loads) feed 8 WMMAs.
// Optional fused epilogue (bias add + batchnorm) when bias != nullptr.
__global__ __launch_bounds__(256) void gemm_wmma_kernel(
    const f16* __restrict__ A, const f16* __restrict__ Bm, float* __restrict__ Co,
    int Nd, int Kd,
    const float* __restrict__ bias, const float* __restrict__ bng,
    const float* __restrict__ bnb, const float* __restrict__ bnr)
{
  const int lane = threadIdx.x & 31;
  const int wave = threadIdx.x >> 5;
  const int grp  = lane >> 4;
  const int l15  = lane & 15;
  const int m0 = blockIdx.x * 256 + wave * 32;
  const int n0 = blockIdx.y * 64;

  v8f acc[2][4] = {};

  // Running pointers: strength-reduced, all inner-loop addressing becomes
  // immediate offsets on global_load_b128.
  const f16* pa0 = A + (size_t)(m0 + l15) * Kd + grp * 8;
  const f16* pa1 = A + (size_t)(m0 + 16 + l15) * Kd + grp * 8;
  const f16* bp  = Bm + (size_t)l15 + n0;
  const size_t bstep = (size_t)32 * Nd;

  for (int k0 = 0; k0 < Kd; k0 += 32) {
    V16H a0, a1, bfr[4];
    a0.q[0] = *(const uint4*)(pa0);
    a0.q[1] = *(const uint4*)(pa0 + 16);
    a1.q[0] = *(const uint4*)(pa1);
    a1.q[1] = *(const uint4*)(pa1 + 16);
    __builtin_prefetch(pa0 + 64, 0, 3);
    __builtin_prefetch(pa1 + 64, 0, 3);
#pragma unroll
    for (int nt = 0; nt < 4; ++nt) {
#pragma unroll
      for (int s = 0; s < 16; ++s) {
        const int kk = (s < 8) ? grp * 8 + s : 16 + grp * 8 + (s - 8);
        bfr[nt].v[s] = bp[(size_t)kk * Nd + nt * 16];
      }
    }
#pragma unroll
    for (int nt = 0; nt < 4; ++nt) {
      acc[0][nt] = wmma_f16(a0, bfr[nt], acc[0][nt]);
      acc[1][nt] = wmma_f16(a1, bfr[nt], acc[1][nt]);
    }
    pa0 += 32;
    pa1 += 32;
    bp  += bstep;
  }

#pragma unroll
  for (int mt = 0; mt < 2; ++mt)
#pragma unroll
    for (int nt = 0; nt < 4; ++nt) {
      const int n = n0 + nt * 16 + l15;
      float sc = 1.f, badd = 0.f, bsh = 0.f;
      if (bias) {
        sc = rsqrtf(bnr[n] + BN_EPS_F) * bng[n];
        badd = bias[n];
        bsh = bnb[n];
      }
#pragma unroll
      for (int r = 0; r < 8; ++r) {
        const int m = m0 + mt * 16 + grp * 8 + r;
        float v = acc[mt][nt][r];
        if (bias) v = (v + badd) * sc + bsh;
        Co[(size_t)m * Nd + n] = v;
      }
    }
}

// ---------------- Kernel 4a: q layernorm + rope -> f16 (B,H,S,QD), scale folded ----------------
__global__ __launch_bounds__(128) void qnorm_rope_kernel(
    const float* __restrict__ qkv, const float* __restrict__ g,
    const float* __restrict__ be, f16* __restrict__ qf)
{
  const int lane = threadIdx.x & 31;
  const int wid = blockIdx.x * 4 + (threadIdx.x >> 5);   // (b*S+s)*H + h
  const int h = wid % NHEAD;
  const int row_bs = wid / NHEAD;                         // b*S + s
  const int s = row_bs % NSEQ;
  const int b = row_bs / NSEQ;
  const float* row = qkv + (size_t)row_bs * NNT + h * NQD;

  float xr[4]; float sum = 0.f;
#pragma unroll
  for (int i = 0; i < 4; ++i) { xr[i] = row[lane + 32 * i]; sum += xr[i]; }
  sum = wave_sum32(sum);
  const float mean = sum * (1.f / 128.f);
  float vs = 0.f;
#pragma unroll
  for (int i = 0; i < 4; ++i) { float d = xr[i] - mean; vs += d * d; }
  vs = wave_sum32(vs);
  const float inv = rsqrtf(vs * (1.f / 128.f) + LN_EPS_F);
  float nn[4];
#pragma unroll
  for (int i = 0; i < 4; ++i) {
    int d = lane + 32 * i;
    nn[i] = (xr[i] - mean) * inv * g[d] + be[d];
  }
  f16* dst = qf + ((size_t)(b * NHEAD + h) * NSEQ + s) * NQD;
#pragma unroll
  for (int i = 0; i < 2; ++i) {
    int d = lane + 32 * i;
    float ang = (float)s * __expf(-LN_ROPE * (float)d * (1.f / 64.f));
    float sn, cs; __sincosf(ang, &sn, &cs);
    dst[d]      = (f16)((nn[i] * cs - nn[i + 2] * sn) * ATTN_SCL);
    dst[d + 64] = (f16)((nn[i] * sn + nn[i + 2] * cs) * ATTN_SCL);
  }
}

// ---------------- Kernel 4b: k layernorm + rope -> f16 transposed (B,KD,S) ----------------
__global__ __launch_bounds__(128) void knorm_rope_kernel(
    const float* __restrict__ qkv, const float* __restrict__ g,
    const float* __restrict__ be, f16* __restrict__ kT)
{
  const int lane = threadIdx.x & 31;
  const int wid = blockIdx.x * 4 + (threadIdx.x >> 5);   // b*S + s
  const int s = wid % NSEQ;
  const int b = wid / NSEQ;
  const float* row = qkv + (size_t)wid * NNT + NHEAD * NQD;

  float xr[4]; float sum = 0.f;
#pragma unroll
  for (int i = 0; i < 4; ++i) { xr[i] = row[lane + 32 * i]; sum += xr[i]; }
  sum = wave_sum32(sum);
  const float mean = sum * (1.f / 128.f);
  float vs = 0.f;
#pragma unroll
  for (int i = 0; i < 4; ++i) { float d = xr[i] - mean; vs += d * d; }
  vs = wave_sum32(vs);
  const float inv = rsqrtf(vs * (1.f / 128.f) + LN_EPS_F);
  float nn[4];
#pragma unroll
  for (int i = 0; i < 4; ++i) {
    int d = lane + 32 * i;
    nn[i] = (xr[i] - mean) * inv * g[d] + be[d];
  }
#pragma unroll
  for (int i = 0; i < 2; ++i) {
    int d = lane + 32 * i;
    float ang = (float)s * __expf(-LN_ROPE * (float)d * (1.f / 64.f));
    float sn, cs; __sincosf(ang, &sn, &cs);
    kT[((size_t)b * NKD + d) * NSEQ + s]      = (f16)(nn[i] * cs - nn[i + 2] * sn);
    kT[((size_t)b * NKD + d + 64) * NSEQ + s] = (f16)(nn[i] * sn + nn[i + 2] * cs);
  }
}

// ---------------- Kernel 4c: v layernorm + rope -> f16 (B,S,VD) ----------------
__global__ __launch_bounds__(128) void vnorm_rope_kernel(
    const float* __restrict__ qkv, const float* __restrict__ g,
    const float* __restrict__ be, f16* __restrict__ vo)
{
  const int lane = threadIdx.x & 31;
  const int wid = blockIdx.x * 4 + (threadIdx.x >> 5);   // b*S + s
  const int s = wid % NSEQ;
  const float* row = qkv + (size_t)wid * NNT + NHEAD * NQD + NKD;

  float xr[6]; float sum = 0.f;
#pragma unroll
  for (int i = 0; i < 6; ++i) { xr[i] = row[lane + 32 * i]; sum += xr[i]; }
  sum = wave_sum32(sum);
  const float mean = sum * (1.f / 192.f);
  float vs = 0.f;
#pragma unroll
  for (int i = 0; i < 6; ++i) { float d = xr[i] - mean; vs += d * d; }
  vs = wave_sum32(vs);
  const float inv = rsqrtf(vs * (1.f / 192.f) + LN_EPS_F);
  float nn[6];
#pragma unroll
  for (int i = 0; i < 6; ++i) {
    int d = lane + 32 * i;
    nn[i] = (xr[i] - mean) * inv * g[d] + be[d];
  }
  f16* dst = vo + (size_t)wid * NVD;
#pragma unroll
  for (int i = 0; i < 3; ++i) {
    int d = lane + 32 * i;
    float ang = (float)s * __expf(-LN_ROPE * (float)d * (1.f / 96.f));
    float sn, cs; __sincosf(ang, &sn, &cs);
    dst[d]      = (f16)(nn[i] * cs - nn[i + 3] * sn);
    dst[d + 96] = (f16)(nn[i] * sn + nn[i + 3] * cs);
  }
}

// ---------------- Kernel 5: flash attention (one wave = 16 q rows) ----------------
__global__ __launch_bounds__(256) void flash_attn_kernel(
    const f16* __restrict__ qf, const f16* __restrict__ kT,
    const f16* __restrict__ vf, f16* __restrict__ of)
{
  __shared__ f16 plds[8 * 512];   // per-wave 16x32 P staging tile
  const int lane = threadIdx.x & 31;
  const int wave = threadIdx.x >> 5;
  const int grp  = lane >> 4;
  const int l15  = lane & 15;
  const int b = blockIdx.x / NHEAD;
  const int h = blockIdx.x % NHEAD;
  const int qbase = blockIdx.y * 128 + wave * 16;

  // Load Q A-fragments: 16 rows x 128 (four K-chunks of 32)
  V16H qa[4];
  {
    const f16* qp = qf + ((size_t)(b * NHEAD + h) * NSEQ + qbase + l15) * NQD;
#pragma unroll
    for (int c = 0; c < 4; ++c) {
      qa[c].q[0] = *(const uint4*)(qp + c * 32 + grp * 8);
      qa[c].q[1] = *(const uint4*)(qp + c * 32 + 16 + grp * 8);
    }
  }

  v8f oacc[12] = {};
  float mrow[8], lrow[8];
#pragma unroll
  for (int r = 0; r < 8; ++r) { mrow[r] = -3.0e38f; lrow[r] = 0.f; }

  // Running pointers: per-iteration addressing folds into 24-bit ioffsets.
  const f16* kpl  = kT + (size_t)b * NKD * NSEQ + (size_t)l15; // += 32 per step
  const f16* vrow = vf + (size_t)b * NSEQ * NVD + (size_t)l15;  // += 32*NVD
  f16* mylds = plds + wave * 512;

  for (int j = 0; j < NSEQ; j += 32) {
    // ---- S = Q * K^T : 16 x 32 logits (two 16x16 C-tiles) ----
    v8f s0 = {}, s1 = {};
#pragma unroll
    for (int c = 0; c < 4; ++c) {
      V16H kb0, kb1;
      const f16* kp = kpl + c * (32 * NSEQ);
#pragma unroll
      for (int s = 0; s < 16; ++s) {
        const int kk = (s < 8) ? grp * 8 + s : 16 + grp * 8 + (s - 8);
        kb0.v[s] = kp[(size_t)kk * NSEQ];
        kb1.v[s] = kp[(size_t)kk * NSEQ + 16];
      }
      s0 = wmma_f16(qa[c], kb0, s0);
      s1 = wmma_f16(qa[c], kb1, s1);
    }
    // ---- online softmax over this 32-key block ----
    float alpha[8];
#pragma unroll
    for (int r = 0; r < 8; ++r) {
      float mx = fmaxf(s0[r], s1[r]);
#pragma unroll
      for (int off = 8; off >= 1; off >>= 1) mx = fmaxf(mx, __shfl_xor(mx, off, 16));
      const float mnew = fmaxf(mrow[r], mx);
      alpha[r] = __expf(mrow[r] - mnew);
      const float p0 = __expf(s0[r] - mnew);
      const float p1 = __expf(s1[r] - mnew);
      s0[r] = p0; s1[r] = p1;
      float ls = p0 + p1;
#pragma unroll
      for (int off = 8; off >= 1; off >>= 1) ls += __shfl_xor(ls, off, 16);
      lrow[r] = lrow[r] * alpha[r] + ls;
      mrow[r] = mnew;
    }
#pragma unroll
    for (int t = 0; t < 12; ++t)
#pragma unroll
      for (int r = 0; r < 8; ++r) oacc[t][r] *= alpha[r];

    // ---- transpose P from C-layout to A-layout through LDS ----
#pragma unroll
    for (int r = 0; r < 8; ++r) {
      const int row = r + 8 * grp;
      mylds[row * 32 + l15]      = (f16)s0[r];
      mylds[row * 32 + 16 + l15] = (f16)s1[r];
    }
    __syncthreads();
    V16H pa;
    {
      const f16* pp = mylds + l15 * 32 + grp * 8;
      pa.q[0] = *(const uint4*)(pp);
      pa.q[1] = *(const uint4*)(pp + 16);
    }
    __syncthreads();
    // ---- O += P * V : 12 tiles of 16x16 over VD=192 ----
#pragma unroll
    for (int t = 0; t < 12; ++t) {
      V16H vv;
      const f16* vp = vrow + t * 16;
#pragma unroll
      for (int s = 0; s < 16; ++s) {
        const int kk = (s < 8) ? grp * 8 + s : 16 + grp * 8 + (s - 8);
        vv.v[s] = vp[(size_t)kk * NVD];
      }
      oacc[t] = wmma_f16(pa, vv, oacc[t]);
    }
    kpl  += 32;
    vrow += (size_t)32 * NVD;
  }

  float invl[8];
#pragma unroll
  for (int r = 0; r < 8; ++r) invl[r] = 1.0f / lrow[r];
#pragma unroll
  for (int t = 0; t < 12; ++t)
#pragma unroll
    for (int r = 0; r < 8; ++r) {
      const int row = qbase + r + 8 * grp;
      of[((size_t)b * NSEQ + row) * NHV + h * NVD + t * 16 + l15] =
          (f16)(oacc[t][r] * invl[r]);
    }
}

// ---------------- host launch ----------------
extern "C" void kernel_launch(void* const* d_in, const int* in_sizes, int n_in,
                              void* d_out, int out_size, void* d_ws, size_t ws_size,
                              hipStream_t stream)
{
  (void)in_sizes; (void)n_in; (void)out_size; (void)ws_size;
  const float* x     = (const float*)d_in[0];
  const float* rms_g = (const float*)d_in[1];
  const float* rms_b = (const float*)d_in[2];
  const float* rms_r = (const float*)d_in[3];
  const float* Wq    = (const float*)d_in[4];
  const float* Wk    = (const float*)d_in[5];
  const float* Wv    = (const float*)d_in[6];
  const float* q_g   = (const float*)d_in[7];
  const float* q_b   = (const float*)d_in[8];
  const float* k_g   = (const float*)d_in[9];
  const float* k_b   = (const float*)d_in[10];
  const float* v_g   = (const float*)d_in[11];
  const float* v_b   = (const float*)d_in[12];
  const float* Wo    = (const float*)d_in[13];
  const float* bo    = (const float*)d_in[14];
  const float* on_g  = (const float*)d_in[15];
  const float* on_b  = (const float*)d_in[16];
  const float* on_r  = (const float*)d_in[17];
  float* out = (float*)d_out;

  char* ws = (char*)d_ws;
  auto alloc = [&](size_t bytes) -> char* {
    char* p = ws;
    ws += (bytes + 255) & ~(size_t)255;
    return p;
  };
  f16*   xn   = (f16*)alloc((size_t)NM * NCH * 2);          // ~25 MB
  f16*   WT   = (f16*)alloc((size_t)NCH * NNT * 2);         // ~4 MB
  f16*   WoT  = (f16*)alloc((size_t)NHV * NCH * 2);         // ~4.7 MB
  float* qkv  = (float*)alloc((size_t)NM * NNT * 4);        // ~44 MB
  f16*   qf   = (f16*)alloc((size_t)NBATCH * NHEAD * NSEQ * NQD * 2); // ~16.8 MB
  f16*   kTf  = (f16*)alloc((size_t)NBATCH * NKD * NSEQ * 2);         // ~2 MB
  f16*   vf   = (f16*)alloc((size_t)NBATCH * NSEQ * NVD * 2);         // ~3.1 MB
  f16*   of   = (f16*)alloc((size_t)NM * NHV * 2);                    // ~25 MB

  // 1) input batchnorm -> f16
  {
    size_t threads = (size_t)NM * NCH / 4;
    bn_in_f16_kernel<<<dim3((unsigned)((threads + 255) / 256)), dim3(256), 0, stream>>>(
        x, rms_g, rms_b, rms_r, xn);
  }
  // 2) weight transposes -> f16
  {
    size_t e1 = (size_t)NCH * NNT;
    wqkv_t_kernel<<<dim3((unsigned)((e1 + 255) / 256)), dim3(256), 0, stream>>>(Wq, Wk, Wv, WT);
    size_t e2 = (size_t)NHV * NCH;
    wo_t_kernel<<<dim3((unsigned)((e2 + 255) / 256)), dim3(256), 0, stream>>>(Wo, WoT);
  }
  // 3) fused QKV projection: (8192 x 1536) * (1536 x 1344) -> f32
  gemm_wmma_kernel<<<dim3(NM / 256, NNT / 64), dim3(256), 0, stream>>>(
      xn, WT, qkv, NNT, NCH, nullptr, nullptr, nullptr, nullptr);
  // 4) layernorm + rope for q / k / v
  qnorm_rope_kernel<<<dim3(NBATCH * NSEQ * NHEAD / 4), dim3(128), 0, stream>>>(qkv, q_g, q_b, qf);
  knorm_rope_kernel<<<dim3(NBATCH * NSEQ / 4), dim3(128), 0, stream>>>(qkv, k_g, k_b, kTf);
  vnorm_rope_kernel<<<dim3(NBATCH * NSEQ / 4), dim3(128), 0, stream>>>(qkv, v_g, v_b, vf);
  // 5) flash attention -> o (f16, B,S,H*VD)
  flash_attn_kernel<<<dim3(NBATCH * NHEAD, NSEQ / 128), dim3(256), 0, stream>>>(qf, kTf, vf, of);
  // 6) output projection + bias + batchnorm -> f32 out
  gemm_wmma_kernel<<<dim3(NM / 256, NCH / 64), dim3(256), 0, stream>>>(
      of, WoT, out, NCH, NHV, bo, on_g, on_b, on_r);
}
